// DeepState_56478819942438
// MI455X (gfx1250) — compile-verified, matrix operand order checked by reading, not run on hardware
//
#include <hip/hip_runtime.h>

// ---------------------------------------------------------------------------
// Problem constants: B=256, S=512, IN=32, H=256, D=32, STATE=4, PRED=96.
// ---------------------------------------------------------------------------
static constexpr int kB    = 256;
static constexpr int kS    = 512;
static constexpr int kIN   = 32;
static constexpr int kH    = 256;
static constexpr int kD    = 32;
static constexpr int kST   = 4;
static constexpr int kPRED = 96;
static constexpr int kG    = 3 * kH;    // 768
static constexpr int kPAD  = 264;       // LDS row stride in halves (bank-friendly)

typedef __attribute__((ext_vector_type(16))) _Float16 v16h;
typedef __attribute__((ext_vector_type(8)))  _Float16 v8h;
typedef __attribute__((ext_vector_type(8)))  float    v8f;

// ---------------------------------------------------------------------------
// WMMA fragment helpers (CDNA5 wave32 16x16x32 f16 layouts, ISA 7.12.2).
//  A: lane L holds row m = L&15; element i -> K = (i&7) + 16*(i>>3) + 8*(L>>4)
//  B: pre-packed so each lane reads 16 contiguous halves (32-byte load).
// ---------------------------------------------------------------------------
__device__ __forceinline__ v16h make_a(const _Float16* p) {
  v8h lo = *(const v8h*)(p);
  v8h hi = *(const v8h*)(p + 16);
  v16h a;
#pragma unroll
  for (int i = 0; i < 8; ++i) { a[i] = lo[i]; a[i + 8] = hi[i]; }
  return a;
}

__device__ __forceinline__ v16h load_b_frag(const _Float16* __restrict__ pack,
                                            int nb, int ksteps, int kk,
                                            int lane) {
  return *(const v16h*)(pack + (((size_t)nb * ksteps + kk) * 32 + lane) * 16);
}

__device__ __forceinline__ float sigf(float x) {
  return 1.0f / (1.0f + expf(-x));
}

#define WMMA(acc, a, b) \
  acc = __builtin_amdgcn_wmma_f32_16x16x32_f16(false, a, false, b, (short)0, acc, false, false)

// Prevent LICM from hoisting loop-invariant weight loads into (spilled)
// registers: with this clobber the compiler must re-issue the L2 loads each
// timestep instead of round-tripping them through scratch.
#define NO_HOIST() asm volatile("" ::: "memory")

// ---------------------------------------------------------------------------
// Pack row-major f32 W[N][K] into f16 WMMA B fragments [nb][kk][lane][i].
// ---------------------------------------------------------------------------
__global__ void pack_weights_kernel(const float* __restrict__ W,
                                    _Float16* __restrict__ out, int N, int K) {
  int idx = blockIdx.x * blockDim.x + threadIdx.x;
  if (idx >= N * K) return;
  int ksteps = K >> 5;
  int i    = idx & 15;
  int lane = (idx >> 4) & 31;
  int rem  = idx >> 9;
  int kk   = rem % ksteps;
  int nb   = rem / ksteps;
  int n = nb * 16 + (lane & 15);
  int k = kk * 32 + (i & 7) + ((i >> 3) << 4) + ((lane >> 4) << 3);
  out[idx] = (_Float16)W[(size_t)n * K + k];
}

__global__ void cvt_f32_f16_kernel(const float* __restrict__ in,
                                   _Float16* __restrict__ out, int n) {
  int idx = blockIdx.x * blockDim.x + threadIdx.x;
  if (idx < n) out[idx] = (_Float16)in[idx];
}

// ---------------------------------------------------------------------------
// One GRU layer scan for this block's 16 batch rows.  h (f16) lives in LDS,
// double-buffered; each wave keeps the f32 copy of its own 16x16 tile in
// registers across all timesteps.  One workgroup barrier per step.
// ---------------------------------------------------------------------------
template <int XK, bool WRITE_SEQ>
__device__ __forceinline__ void gru_phase(
    const _Float16* __restrict__ xbase,   // + m*xrstride + t*xtstride + k
    long long xrstride, int xtstride,
    const _Float16* __restrict__ wih,     // [48][XK][32][16]
    const _Float16* __restrict__ whh,     // [48][8][32][16]
    const float* __restrict__ b_ih, const float* __restrict__ b_hh,
    _Float16* __restrict__ seqbase,       // + lrow*seqrstride + t*kH + col
    long long seqrstride,
    _Float16 (*hbuf)[16][kPAD],           // shared double buffer
    int wid, int lane) {
  const int ct  = wid;
  const int m   = lane & 15;
  const int kg  = (lane >> 4) << 3;
  const int rl0 = (lane >> 4) << 3;
  const int col = ct * 16 + (lane & 15);

  const float bir = b_ih[col],          bhr = b_hh[col];
  const float biz = b_ih[col + kH],     bhz = b_hh[col + kH];
  const float bin = b_ih[col + 2 * kH], bhn = b_hh[col + 2 * kH];

  // For the tiny layer-0 input weights (XK==1, 3 fragments = 24 VGPRs) we
  // cache the fragments as SSA values: they survive the clobber below.
  v16h wih_r, wih_z, wih_n;
  if (XK == 1) {
    wih_r = load_b_frag(wih, ct,      1, 0, lane);
    wih_z = load_b_frag(wih, ct + 16, 1, 0, lane);
    wih_n = load_b_frag(wih, ct + 32, 1, 0, lane);
  }

  v8f hreg = {};   // f32 master copy of this wave's 16x16 h tile
  int cur = 0;

  for (int t = 0; t < kS; ++t) {
    NO_HOIST();    // keep weight loads streaming from L2, not scratch
    v8f c_r = {}, c_z = {}, c_hn = {}, c_xn = {};

    // recurrent part: K = 256 -> 8 k-steps, A from LDS
#pragma unroll
    for (int kk = 0; kk < 8; ++kk) {
      v16h a  = make_a(&hbuf[cur][m][kk * 32 + kg]);
      v16h br = load_b_frag(whh, ct,      8, kk, lane);
      v16h bz = load_b_frag(whh, ct + 16, 8, kk, lane);
      v16h bn = load_b_frag(whh, ct + 32, 8, kk, lane);
      WMMA(c_r, a, br);
      WMMA(c_z, a, bz);
      WMMA(c_hn, a, bn);
    }
    // input part: A streamed from global
    if (XK == 1) {
      v16h a = make_a(xbase + (size_t)m * xrstride +
                      (size_t)t * xtstride + kg);
      WMMA(c_r, a, wih_r);
      WMMA(c_z, a, wih_z);
      WMMA(c_xn, a, wih_n);
    } else {
#pragma unroll
      for (int kk = 0; kk < XK; ++kk) {
        v16h a  = make_a(xbase + (size_t)m * xrstride +
                         (size_t)t * xtstride + kk * 32 + kg);
        v16h br = load_b_frag(wih, ct,      XK, kk, lane);
        v16h bz = load_b_frag(wih, ct + 16, XK, kk, lane);
        v16h bn = load_b_frag(wih, ct + 32, XK, kk, lane);
        WMMA(c_r, a, br);
        WMMA(c_z, a, bz);
        WMMA(c_xn, a, bn);
      }
    }

#pragma unroll
    for (int rr = 0; rr < 8; ++rr) {
      float r  = sigf(c_r[rr] + bir + bhr);
      float z  = sigf(c_z[rr] + biz + bhz);
      float nn = tanhf(c_xn[rr] + bin + r * (c_hn[rr] + bhn));
      float hv = (1.0f - z) * nn + z * hreg[rr];
      hreg[rr] = hv;
      _Float16 h16v = (_Float16)hv;
      hbuf[cur ^ 1][rl0 + rr][col] = h16v;
      if (WRITE_SEQ)
        seqbase[(size_t)(rl0 + rr) * seqrstride + (size_t)t * kH + col] = h16v;
    }
    __syncthreads();
    cur ^= 1;
  }
  // kS is even -> final h sits in hbuf[0].
}

// ---------------------------------------------------------------------------
// Fused persistent kernel: 16 blocks x 512 threads (16 waves).  Block b owns
// batch rows [16b, 16b+16): layer0 scan -> layer1 scan -> projection ->
// 96-step rollout, all with workgroup-local synchronization only.
// ---------------------------------------------------------------------------
__global__ __launch_bounds__(512) void deepstate_fused_kernel(
    const _Float16* __restrict__ x16,     // [B][S][IN] f16
    const _Float16* __restrict__ wih0p, const _Float16* __restrict__ whh0p,
    const float* __restrict__ bih0, const float* __restrict__ bhh0,
    const _Float16* __restrict__ wih1p, const _Float16* __restrict__ whh1p,
    const float* __restrict__ bih1, const float* __restrict__ bhh1,
    const _Float16* __restrict__ wpp,     // packed W_proj [8][8][32][16]
    const float* __restrict__ bproj,
    const float* __restrict__ Cm,
    const float* __restrict__ rlvl, const float* __restrict__ rtrd,
    const float* __restrict__ rgam, const float* __restrict__ omga,
    _Float16* __restrict__ h1seq,         // [B][S][H] f16 scratch
    float* __restrict__ out) {            // [B][PRED][D]
  __shared__ _Float16 hbuf[2][16][kPAD];
  __shared__ float    projbuf[16][kST * kD];

  const int wid  = threadIdx.x >> 5;
  const int lane = threadIdx.x & 31;
  const int row0 = blockIdx.x * 16;

  // zero the initial h buffer
  {
    _Float16* z = &hbuf[0][0][0];
    for (int i = threadIdx.x; i < 16 * kPAD; i += 512) z[i] = (_Float16)0.0f;
  }
  __syncthreads();

  // ---- layer 0: x (K=32) -> h1seq ----
  gru_phase<1, true>(x16 + (size_t)row0 * kS * kIN, (long long)kS * kIN, kIN,
                     wih0p, whh0p, bih0, bhh0,
                     h1seq + (size_t)row0 * kS * kH, (long long)kS * kH,
                     hbuf, wid, lane);

  // make phase-1 global stores visible before phase-2 reads, reset h
  __threadfence();
  {
    _Float16* z = &hbuf[0][0][0];
    for (int i = threadIdx.x; i < 16 * kPAD; i += 512) z[i] = (_Float16)0.0f;
  }
  __syncthreads();

  // ---- layer 1: h1seq (K=256) -> final h in hbuf[0] ----
  gru_phase<8, false>(h1seq + (size_t)row0 * kS * kH, (long long)kS * kH, kH,
                      wih1p, whh1p, bih1, bhh1,
                      (_Float16*)nullptr, 0LL,
                      hbuf, wid, lane);

  // ---- projection: proj[16][128] = h_final @ W_proj^T + b_proj ----
  if (wid < 8) {   // wave-uniform branch; EXEC all-ones inside
    const int m  = lane & 15;
    const int kg = (lane >> 4) << 3;
    v8f c = {};
#pragma unroll
    for (int kk = 0; kk < 8; ++kk) {
      v16h a = make_a(&hbuf[0][m][kk * 32 + kg]);
      v16h b = load_b_frag(wpp, wid, 8, kk, lane);
      WMMA(c, a, b);
    }
    int col  = wid * 16 + (lane & 15);
    float bv = bproj[col];
    int rl0  = (lane >> 4) << 3;
#pragma unroll
    for (int rr = 0; rr < 8; ++rr)
      projbuf[rl0 + rr][col] = c[rr] + bv;
  }
  __syncthreads();

  // ---- 96-step rollout: 512 threads = 16 rows x 32 d exactly ----
  {
    int lrow = threadIdx.x >> 5;
    int d    = threadIdx.x & 31;
    int grow = row0 + lrow;
    float s0 = projbuf[lrow][d * kST + 0];
    float s1 = projbuf[lrow][d * kST + 1];
    float s2 = projbuf[lrow][d * kST + 2];
    float s3 = projbuf[lrow][d * kST + 3];

    float a_l = sigf(rlvl[d]) * 0.15f + 0.85f;
    float a_t = sigf(rtrd[d]) * 0.25f + 0.70f;
    float g   = sigf(rgam[d]) * 0.20f + 0.80f;
    float cs = cosf(omga[d]), sn = sinf(omga[d]);
    float r00 = g * cs, r01 = -g * sn, r10 = g * sn, r11 = g * cs;
    float C0 = Cm[d * kST + 0], C1 = Cm[d * kST + 1];
    float C2 = Cm[d * kST + 2], C3 = Cm[d * kST + 3];

    for (int p = 0; p < kPRED; ++p) {
      float n0 = s0 * a_l;
      float n1 = s1 * a_t;
      float n2 = s2 * r00 + s3 * r10;
      float n3 = s2 * r01 + s3 * r11;
      s0 = n0; s1 = n1; s2 = n2; s3 = n3;
      out[((size_t)grow * kPRED + p) * kD + d] =
          C0 * n0 + C1 * n1 + C2 * n2 + C3 * n3;
    }
  }
}

// ---------------------------------------------------------------------------
// Host launch: 5 pack + 1 convert + 1 fused persistent kernel.
// ---------------------------------------------------------------------------
extern "C" void kernel_launch(void* const* d_in, const int* in_sizes, int n_in,
                              void* d_out, int out_size, void* d_ws,
                              size_t ws_size, hipStream_t stream) {
  (void)in_sizes; (void)n_in; (void)out_size; (void)ws_size;
  const float* x     = (const float*)d_in[0];
  const float* Wih0  = (const float*)d_in[1];
  const float* Whh0  = (const float*)d_in[2];
  const float* bih0  = (const float*)d_in[3];
  const float* bhh0  = (const float*)d_in[4];
  const float* Wih1  = (const float*)d_in[5];
  const float* Whh1  = (const float*)d_in[6];
  const float* bih1  = (const float*)d_in[7];
  const float* bhh1  = (const float*)d_in[8];
  const float* Wproj = (const float*)d_in[9];
  const float* bproj = (const float*)d_in[10];
  const float* Cm    = (const float*)d_in[11];
  const float* rlvl  = (const float*)d_in[12];
  const float* rtrd  = (const float*)d_in[13];
  const float* rgam  = (const float*)d_in[14];
  const float* omga  = (const float*)d_in[15];
  float* out = (float*)d_out;

  char* ws = (char*)d_ws;
  size_t off = 0;
  auto wsalloc = [&](size_t bytes) -> void* {
    void* p = ws + off;
    off = (off + bytes + 255) & ~(size_t)255;
    return p;
  };
  _Float16* x16   = (_Float16*)wsalloc((size_t)kB * kS * kIN * 2);
  _Float16* h1seq = (_Float16*)wsalloc((size_t)kB * kS * kH * 2);
  _Float16* wih0p = (_Float16*)wsalloc((size_t)kG * kIN * 2);
  _Float16* whh0p = (_Float16*)wsalloc((size_t)kG * kH * 2);
  _Float16* wih1p = (_Float16*)wsalloc((size_t)kG * kH * 2);
  _Float16* whh1p = (_Float16*)wsalloc((size_t)kG * kH * 2);
  _Float16* wpp   = (_Float16*)wsalloc((size_t)(kST * kD) * kH * 2);

  pack_weights_kernel<<<(kG * kIN + 255) / 256, 256, 0, stream>>>(Wih0, wih0p, kG, kIN);
  pack_weights_kernel<<<(kG * kH + 255) / 256, 256, 0, stream>>>(Whh0, whh0p, kG, kH);
  pack_weights_kernel<<<(kG * kH + 255) / 256, 256, 0, stream>>>(Wih1, wih1p, kG, kH);
  pack_weights_kernel<<<(kG * kH + 255) / 256, 256, 0, stream>>>(Whh1, whh1p, kG, kH);
  pack_weights_kernel<<<(kST * kD * kH + 255) / 256, 256, 0, stream>>>(Wproj, wpp, kST * kD, kH);
  int nx = kB * kS * kIN;
  cvt_f32_f16_kernel<<<(nx + 255) / 256, 256, 0, stream>>>(x, x16, nx);

  deepstate_fused_kernel<<<kB / 16, 512, 0, stream>>>(
      x16, wih0p, whh0p, bih0, bhh0, wih1p, whh1p, bih1, bhh1,
      wpp, bproj, Cm, rlvl, rtrd, rgam, omga, h1seq, out);
}
